// FastUpConvolution_34273839022962
// MI455X (gfx1250) — matrix operand order repaired and checked by reading, not compiled, not run on hardware
//
#include <hip/hip_runtime.h>
#include <hip/hip_bf16.h>

typedef __attribute__((ext_vector_type(16))) __bf16 v16bf;
typedef __attribute__((ext_vector_type(8)))  float  v8f;
typedef __attribute__((ext_vector_type(4)))  unsigned int v4u;
typedef __attribute__((ext_vector_type(8)))  int  v8i;
typedef __attribute__((ext_vector_type(4)))  int  v4i;

union Frag16 { v16bf v; uint4 q[2]; };

#if defined(__has_builtin)
#if __has_builtin(__builtin_amdgcn_tensor_load_to_lds) && __has_builtin(__builtin_amdgcn_s_wait_tensorcnt)
#define HAVE_TDM 1
#endif
#endif
#ifndef HAVE_TDM
#define HAVE_TDM 0
#endif

static __device__ __forceinline__ unsigned short f2bf(float f) {
    union { float f; unsigned u; } v; v.f = f;
    unsigned r = v.u + 0x7FFFu + ((v.u >> 16) & 1u);   // round-to-nearest-even
    return (unsigned short)(r >> 16);
}
static __device__ __forceinline__ unsigned pk2(unsigned short a, unsigned short b) {
    return (unsigned)a | ((unsigned)b << 16);
}

// ---------------------------------------------------------------------------
// Kernel 1: pack four conv filter banks into one bf16 [N=128][K=576] matrix
// (zero-padding smaller kernels onto the common 3x3 footprint), pack biases,
// zero BN accumulators.  K = (kh*3+kw)*64 + cin ; n = conv*32 + cout.
// ---------------------------------------------------------------------------
__global__ void upconv_pack(const float* __restrict__ w1, const float* __restrict__ b1,
                            const float* __restrict__ w2, const float* __restrict__ b2,
                            const float* __restrict__ w3, const float* __restrict__ b3,
                            const float* __restrict__ w4, const float* __restrict__ b4,
                            unsigned short* __restrict__ Wpack,
                            float* __restrict__ biasp,
                            float* __restrict__ gsum, float* __restrict__ gsumsq) {
    const int n = threadIdx.x;             // 0..127
    if (n >= 128) return;
    const int conv = n >> 5, cout = n & 31;
    const float* bsrc = (conv == 0) ? b1 : (conv == 1) ? b2 : (conv == 2) ? b3 : b4;
    biasp[n] = bsrc[cout];
    if (n < 32) { gsum[n] = 0.0f; gsumsq[n] = 0.0f; }
    for (int kpos = 0; kpos < 9; ++kpos) {
        const int kh = kpos / 3, kw = kpos % 3;
        for (int cin = 0; cin < 64; ++cin) {
            float v;
            if (conv == 0)      v = w1[((cout * 64 + cin) * 3 + kh) * 3 + kw];
            else if (conv == 1) v = (kh < 2) ? w2[((cout * 64 + cin) * 2 + kh) * 3 + kw] : 0.0f;
            else if (conv == 2) v = (kw < 2) ? w3[((cout * 64 + cin) * 3 + kh) * 2 + kw] : 0.0f;
            else                v = (kh < 2 && kw < 2) ? w4[((cout * 64 + cin) * 2 + kh) * 2 + kw] : 0.0f;
            Wpack[n * 576 + kpos * 64 + cin] = f2bf(v);
        }
    }
}

// ---------------------------------------------------------------------------
// Kernel 2: fused implicit-GEMM conv via v_wmma_f32_16x16x32_bf16.
// Block = one (b,h) row: M = 128 w-positions, N = 128 merged channels,
// K = 576 in chunks of 32.  Double-buffered LDS; weight tiles staged with the
// Tensor Data Mover (TDM) when available; activations converted f32->bf16 on
// the fly with branchless clamped loads.
// ---------------------------------------------------------------------------
__global__ __launch_bounds__(256) void upconv_gemm(
        const float* __restrict__ x,
        const unsigned short* __restrict__ Wpack,
        const float* __restrict__ biasp,
        float* __restrict__ out,
        float* __restrict__ gsum, float* __restrict__ gsumsq) {

    __shared__ __align__(16) unsigned short As[2][128 * 40];  // [buf][m=w][kl], pitch 40 bf16
    __shared__ __align__(16) unsigned short Bs[2][128 * 40];  // [buf][n]  [kl], pitch 40 bf16
    __shared__ float s_sum[32];
    __shared__ float s_sumsq[32];

    const int tid = threadIdx.x;
    const int bh  = blockIdx.x;            // b*128 + h
    const int bi  = bh >> 7;
    const int h   = bh & 127;

    if (tid < 32) { s_sum[tid] = 0.0f; s_sumsq[tid] = 0.0f; }

    // loader role: 256 threads = (khalf 0..1) x (lw 0..127)
    const int lw    = tid & 127;           // w (A loads) / n (B fallback loads)
    const int khalf = tid >> 7;

    // wave role (wave32)
    const int lane  = tid & 31;
    const int l16   = lane & 15;
    const int hi    = lane >> 4;
    const int wid   = tid >> 5;            // 0..7
    const int waveM = wid >> 1;            // rows waveM*32..+31
    const int waveN = wid & 1;             // cols waveN*64..+63

    // ---- B tile staging: TDM descriptor issue (wave 0) or fallback loads ---
    auto stageB = [&](int kk, int buf) {
#if HAVE_TDM
        if (wid == 0) {
            const unsigned ldsOff = (unsigned)(uintptr_t)(void*)&Bs[buf][0];
            const unsigned long long ga =
                (unsigned long long)(uintptr_t)(Wpack + kk * 32);
            // D# group 0: count=1 | lds_addr | global_addr | type=2
            v4u g0 = { 1u, ldsOff, (unsigned)(ga & 0xFFFFFFFFu),
                       (unsigned)(((ga >> 32) & 0x01FFFFFFu) | 0x80000000u) };
            // D# group 1: data_size=2B, pad_enable, pad_interval=16dw, pad_amount=4dw,
            // tensor 576x128 (stride 576), tile 32x128
            const unsigned d0 = 0x10000u | (1u << 20) | (3u << 22) | (3u << 25);
            v8i g1 = { (int)d0, (int)(576u << 16), (int)(128u << 16),
                       (int)(32u << 16), 128, 576, 0, 0 };
            v4i gz = { 0, 0, 0, 0 };
#if defined(__clang_major__) && (__clang_major__ >= 23)
            v8i gz8 = { 0, 0, 0, 0, 0, 0, 0, 0 };
            __builtin_amdgcn_tensor_load_to_lds(g0, g1, gz, gz, gz8, 0);
#else
            __builtin_amdgcn_tensor_load_to_lds(g0, g1, gz, gz, 0);
#endif
        }
#else
        const unsigned short* wp = Wpack + lw * 576 + kk * 32 + khalf * 16;
        uint4 q0 = ((const uint4*)wp)[0];
        uint4 q1 = ((const uint4*)wp)[1];
        *(uint4*)&Bs[buf][lw * 40 + khalf * 16]     = q0;
        *(uint4*)&Bs[buf][lw * 40 + khalf * 16 + 8] = q1;
#endif
    };

    // ---- A tile: branchless clamped global loads -> packed bf16 registers --
    auto loadA = [&](int kk, uint4& q0, uint4& q1) {
        const int kpos = kk >> 1;
        const int kh   = kpos / 3, kw = kpos % 3;
        const int row  = h  + kh - 1;
        const int col  = lw + kw - 1;
        const bool ok  = ((unsigned)row < 128u) && ((unsigned)col < 128u);
        const int rc   = min(max(row, 0), 127);
        const int cc   = min(max(col, 0), 127);
        const int cin0 = ((kk & 1) << 5) + (khalf << 4);
        const float* xp = x + (((size_t)bi * 64 + cin0) << 14) + rc * 128 + cc;
        unsigned short hv[16];
#pragma unroll
        for (int i = 0; i < 16; ++i) {
            float v = xp[(size_t)i << 14];           // always in-bounds
            hv[i] = f2bf(ok ? v : 0.0f);             // v_cndmask, no branches
        }
        q0 = make_uint4(pk2(hv[0], hv[1]),   pk2(hv[2], hv[3]),
                        pk2(hv[4], hv[5]),   pk2(hv[6], hv[7]));
        q1 = make_uint4(pk2(hv[8], hv[9]),   pk2(hv[10], hv[11]),
                        pk2(hv[12], hv[13]), pk2(hv[14], hv[15]));
    };
    auto storeA = [&](int buf, const uint4& q0, const uint4& q1) {
        *(uint4*)&As[buf][lw * 40 + khalf * 16]     = q0;
        *(uint4*)&As[buf][lw * 40 + khalf * 16 + 8] = q1;
    };

    v8f acc[2][4] = {};
    uint4 aq0, aq1;

    // ---- prologue: stage chunk 0 into buffer 0 ----------------------------
    stageB(0, 0);
    loadA(0, aq0, aq1);
    storeA(0, aq0, aq1);

    for (int kk = 0; kk < 18; ++kk) {
        const int cur = kk & 1;
#if HAVE_TDM
        if (wid == 0) __builtin_amdgcn_s_wait_tensorcnt((short)0);
#endif
        __syncthreads();

        const bool more = (kk < 17);
        if (more) {
            stageB(kk + 1, cur ^ 1);        // TDM -> other buffer (async)
            loadA(kk + 1, aq0, aq1);        // global loads overlap compute
        }

        // ---- fragments, contiguous k = 16*hi + s in A and B (from current buffer) ---------
        Frag16 af[2];
#pragma unroll
        for (int t = 0; t < 2; ++t) {
            const int r = waveM * 32 + t * 16 + l16;
            af[t].q[0] = *(const uint4*)&As[cur][r * 40 + hi * 16];
            af[t].q[1] = *(const uint4*)&As[cur][r * 40 + hi * 16 + 8];
        }
        Frag16 bf[4];
#pragma unroll
        for (int t = 0; t < 4; ++t) {
            const int n = waveN * 64 + t * 16 + l16;
            bf[t].q[0] = *(const uint4*)&Bs[cur][n * 40 + hi * 16];
            bf[t].q[1] = *(const uint4*)&Bs[cur][n * 40 + hi * 16 + 8];
        }
#pragma unroll
        for (int tM = 0; tM < 2; ++tM)
#pragma unroll
            for (int tN = 0; tN < 4; ++tN)
                acc[tM][tN] = __builtin_amdgcn_wmma_f32_16x16x32_bf16(
                    false, af[tM].v, false, bf[tN].v,
                    (short)0, acc[tM][tN], false, false);

        if (more) storeA(cur ^ 1, aq0, aq1);
    }

    // ---- epilogue: bias, interleaved scatter, BN partial sums -------------
#pragma unroll
    for (int tN = 0; tN < 4; ++tN) {
        const int n    = waveN * 64 + tN * 16 + l16;
        const int conv = n >> 5;
        const int c    = n & 31;
        const int rowp = conv & 1;          // out1:(0,0) out2:(1,0) out3:(0,1) out4:(1,1)
        const int colp = conv >> 1;
        const float bia = biasp[n];
        const size_t obase =
            (((size_t)(bi * 32 + c) * 256) + (size_t)(2 * h + rowp)) * 256 + colp;
        float s = 0.0f, s2 = 0.0f;
#pragma unroll
        for (int tM = 0; tM < 2; ++tM) {
            const int w0 = waveM * 32 + tM * 16 + hi * 8;   // D: M = r + hi*8
#pragma unroll
            for (int r = 0; r < 8; ++r) {
                const float v = acc[tM][tN][r] + bia;
                out[obase + (size_t)2 * (w0 + r)] = v;
                s += v; s2 += v * v;
            }
        }
        __hip_atomic_fetch_add(&s_sum[c],   s,  __ATOMIC_RELAXED, __HIP_MEMORY_SCOPE_WORKGROUP);
        __hip_atomic_fetch_add(&s_sumsq[c], s2, __ATOMIC_RELAXED, __HIP_MEMORY_SCOPE_WORKGROUP);
    }
    __syncthreads();
    if (tid < 32) {
        __hip_atomic_fetch_add(&gsum[tid],   s_sum[tid],   __ATOMIC_RELAXED, __HIP_MEMORY_SCOPE_AGENT);
        __hip_atomic_fetch_add(&gsumsq[tid], s_sumsq[tid], __ATOMIC_RELAXED, __HIP_MEMORY_SCOPE_AGENT);
    }
}

// ---------------------------------------------------------------------------
// Kernel 3: per-channel BN scale/shift from accumulated moments.
// ---------------------------------------------------------------------------
__global__ void upconv_finalize(const float* __restrict__ gsum, const float* __restrict__ gsumsq,
                                const float* __restrict__ gamma, const float* __restrict__ beta,
                                float* __restrict__ scale, float* __restrict__ shiftp) {
    const int c = threadIdx.x;
    if (c >= 32) return;
    const float inv = 1.0f / 2097152.0f;   // B * 2H * 2W
    const float mean = gsum[c] * inv;
    const float var  = gsumsq[c] * inv - mean * mean;
    const float sc   = gamma[c] * rsqrtf(var + 1e-5f);
    scale[c]  = sc;
    shiftp[c] = beta[c] - mean * sc;
}

// ---------------------------------------------------------------------------
// Kernel 4: in-place normalize, float4 vectorized.  Plane (b,c) = 16384
// float4s -> c = (i>>14)&31.
// ---------------------------------------------------------------------------
__global__ __launch_bounds__(256) void upconv_bn(float4* __restrict__ y,
                                                 const float* __restrict__ scale,
                                                 const float* __restrict__ shiftp) {
    const size_t i = (size_t)blockIdx.x * 256 + threadIdx.x;
    const int c = (int)((i >> 14) & 31);
    const float sc = scale[c], sh = shiftp[c];
    float4 v = y[i];
    v.x = v.x * sc + sh;
    v.y = v.y * sc + sh;
    v.z = v.z * sc + sh;
    v.w = v.w * sc + sh;
    y[i] = v;
}

// ---------------------------------------------------------------------------
extern "C" void kernel_launch(void* const* d_in, const int* in_sizes, int n_in,
                              void* d_out, int out_size, void* d_ws, size_t ws_size,
                              hipStream_t stream) {
    const float* x     = (const float*)d_in[0];
    const float* w1    = (const float*)d_in[1];
    const float* b1    = (const float*)d_in[2];
    const float* w2    = (const float*)d_in[3];
    const float* b2    = (const float*)d_in[4];
    const float* w3    = (const float*)d_in[5];
    const float* b3    = (const float*)d_in[6];
    const float* w4    = (const float*)d_in[7];
    const float* b4    = (const float*)d_in[8];
    const float* gamma = (const float*)d_in[9];
    const float* beta  = (const float*)d_in[10];
    float* out = (float*)d_out;

    char* ws = (char*)d_ws;
    unsigned short* Wpack = (unsigned short*)ws;            // 128*576 bf16 = 147456 B
    float* biasp  = (float*)(ws + 147456);                  // 128 f32
    float* gsum   = (float*)(ws + 147968);                  // 32 f32
    float* gsumsq = gsum + 32;
    float* scale  = gsum + 64;
    float* shiftp = gsum + 96;

    upconv_pack<<<1, 128, 0, stream>>>(w1, b1, w2, b2, w3, b3, w4, b4,
                                       Wpack, biasp, gsum, gsumsq);
    upconv_gemm<<<32 * 128, 256, 0, stream>>>(x, Wpack, biasp, out, gsum, gsumsq);
    upconv_finalize<<<1, 32, 0, stream>>>(gsum, gsumsq, gamma, beta, scale, shiftp);
    upconv_bn<<<65536, 256, 0, stream>>>((float4*)out, scale, shiftp);
}
